// PIRGAT_59657095741764
// MI455X (gfx1250) — matrix operand order repaired and checked by reading, compiled, not run on hardware
//
#include <hip/hip_runtime.h>
#include <hip/hip_bf16.h>

typedef _Float16 h16;
typedef __attribute__((ext_vector_type(16))) _Float16 v16h;
typedef __attribute__((ext_vector_type(8)))  _Float16 v8h;
typedef __attribute__((ext_vector_type(8)))  float    v8f;

#define NN  50000
#define EE  800000
#define RRk 8
#define BBk 4
#define NHH 2

static inline unsigned gdiv(long n, long b) { return (unsigned)((n + b - 1) / b); }

// ---------------- generic ----------------
__global__ void k_fill(float* __restrict__ p, float v, long n) {
  long i = (long)blockIdx.x * blockDim.x + threadIdx.x;
  if (i < n) p[i] = v;
}

__global__ void k_cvt(const float* __restrict__ in, h16* __restrict__ out, long n, int relu) {
  long i = (long)blockIdx.x * blockDim.x + threadIdx.x;
  if (i < n) {
    float v = in[i];
    if (relu) v = fmaxf(v, 0.0f);
    out[i] = (h16)v;
  }
}

// ---------------- WMMA GEMM: C[M,ncols] = A[M,64] * Bt[ncols,64]^T ----------------
// One wave32 per 16x16 C tile. K=64 as two K=32 WMMA steps.
// A fragment (16-bit A 16x32, ISA 7.12.2): lane half h, element t -> K = 16*h + t (contiguous like B, not the ISA table order)
// B fragment: lane half h, element t -> K = 16*h + t, N = lane&15 (contiguous in Bt row)
__global__ void __launch_bounds__(32) k_gemm(const h16* __restrict__ A,
                                             const h16* __restrict__ Bt,
                                             float* __restrict__ C, int ncols) {
  int lane = threadIdx.x;
  int hi = lane >> 4;      // lane half
  int lm = lane & 15;
  int row = blockIdx.y * 16 + lm;
  int col = blockIdx.x * 16 + lm;
  v8f acc = {};
  const h16* arow = A + (size_t)row * 64 + 16 * hi;
  const h16* brow = Bt + (size_t)col * 64 + 16 * hi;
#pragma unroll
  for (int kb = 0; kb < 64; kb += 32) {
    v8h a0 = *(const v8h*)(arow + kb);        // K = kb + 16*hi + 0..7
    v8h a1 = *(const v8h*)(arow + kb + 8);    // K = kb + 16*hi + 8..15
    v16h a = __builtin_shufflevector(a0, a1, 0,1,2,3,4,5,6,7,8,9,10,11,12,13,14,15);
    v16h b = *(const v16h*)(brow + kb);       // K = kb + 16*hi + 0..15
    acc = __builtin_amdgcn_wmma_f32_16x16x32_f16(false, a, false, b, (short)0, acc,
                                                 false, false);
  }
  // C layout: VGPR v, lane -> M = v + 8*hi, N = lm
  float* cp = C + (size_t)(blockIdx.y * 16 + 8 * hi) * ncols + blockIdx.x * 16 + lm;
#pragma unroll
  for (int v = 0; v < 8; ++v) cp[(size_t)v * ncols] = acc[v];
}

// ---------------- RGCN ----------------
// Build transposed f16 weights: rows [0, R*dout) = basis-composed relation weights,
// rows [R*dout, (R+1)*dout) = root weight.  Wt[row][i] over i in [0,64).
__global__ void k_wrgcn(const float* __restrict__ rb, const float* __restrict__ rc,
                        const float* __restrict__ rr, h16* __restrict__ Wt, int dout) {
  int rt = (RRk + 1) * dout;
  int i = blockIdx.x * blockDim.x + threadIdx.x;
  if (i >= rt * 64) return;
  int row = i >> 6, c = i & 63;
  float v;
  if (row < RRk * dout) {
    int r = row / dout, o = row - r * dout;
    v = 0.0f;
#pragma unroll
    for (int b = 0; b < BBk; ++b)
      v += rc[r * BBk + b] * rb[((size_t)b * 64 + c) * dout + o];
  } else {
    int o = row - RRk * dout;
    v = rr[(size_t)c * dout + o];
  }
  Wt[(size_t)row * 64 + c] = (h16)v;
}

__global__ void k_rcount(const int* __restrict__ dst, const int* __restrict__ et,
                         float* __restrict__ cnt) {
  int e = blockIdx.x * blockDim.x + threadIdx.x;
  if (e >= EE) return;
  atomicAdd(&cnt[(size_t)dst[e] * RRk + et[e]], 1.0f);
}

// agg = root-block of H + bias
__global__ void k_rinit(const float* __restrict__ H, const float* __restrict__ bias,
                        float* __restrict__ agg, int dout) {
  long i = (long)blockIdx.x * blockDim.x + threadIdx.x;
  if (i >= (long)NN * dout) return;
  int o = (int)(i % dout);
  long n = i / dout;
  int hc = (RRk + 1) * dout;
  agg[i] = H[n * hc + RRk * dout + o] + bias[o];
}

// per-(edge, 4 outputs): agg[dst] += H[src, etype] / max(cnt,1)
__global__ void k_rscatter(const float* __restrict__ H, const int* __restrict__ src,
                           const int* __restrict__ dst, const int* __restrict__ et,
                           const float* __restrict__ cnt, float* __restrict__ agg, int dout) {
  int g = dout >> 2;
  long i = (long)blockIdx.x * blockDim.x + threadIdx.x;
  if (i >= (long)EE * g) return;
  int e = (int)(i / g);
  int go = ((int)(i - (long)e * g)) << 2;
  int d = dst[e], r = et[e], s = src[e];
  int hc = (RRk + 1) * dout;
  float inv = 1.0f / fmaxf(cnt[(size_t)d * RRk + r], 1.0f);
  const float4 hv = *reinterpret_cast<const float4*>(H + (size_t)s * hc + r * dout + go);
  float* ap = agg + (size_t)d * dout + go;
  atomicAdd(ap + 0, hv.x * inv);
  atomicAdd(ap + 1, hv.y * inv);
  atomicAdd(ap + 2, hv.z * inv);
  atomicAdd(ap + 3, hv.w * inv);
}

// ---------------- GAT ----------------
__global__ void k_wgat(const float* __restrict__ gw, h16* __restrict__ Wt, int nhc) {
  int i = blockIdx.x * blockDim.x + threadIdx.x;
  if (i >= nhc * 64) return;
  int row = i >> 6, c = i & 63;  // row = output column, c = input dim
  Wt[i] = (h16)gw[(size_t)c * nhc + row];
}

__global__ void k_glogit(const float* __restrict__ h, const float* __restrict__ gas,
                         const float* __restrict__ gad, float* __restrict__ as_,
                         float* __restrict__ ad_, int C) {
  int i = blockIdx.x * blockDim.x + threadIdx.x;
  if (i >= NN * NHH) return;
  int hh = i & (NHH - 1);
  int n = i >> 1;
  const float* hp = h + (size_t)n * NHH * C + (size_t)hh * C;
  const float* ps = gas + hh * C;
  const float* pd = gad + hh * C;
  float s1 = 0.0f, s2 = 0.0f;
  for (int c = 0; c < C; ++c) { float v = hp[c]; s1 += v * ps[c]; s2 += v * pd[c]; }
  as_[i] = s1;
  ad_[i] = s2;
}

__device__ __forceinline__ void atomicMaxF(float* a, float v) {
  if (v >= 0.0f) atomicMax((int*)a, __float_as_int(v));
  else           atomicMin((unsigned int*)a, __float_as_uint(v));
}

__global__ void k_gedge1(const int* __restrict__ src, const int* __restrict__ dst,
                         const float* __restrict__ as_, const float* __restrict__ ad_,
                         float* __restrict__ aw, float* __restrict__ m) {
  int e = blockIdx.x * blockDim.x + threadIdx.x;
  if (e >= EE + NN) return;
  int s, d;
  if (e < EE) { s = src[e]; d = dst[e]; } else { s = d = e - EE; }
#pragma unroll
  for (int hh = 0; hh < NHH; ++hh) {
    float a = as_[s * NHH + hh] + ad_[d * NHH + hh];
    a = a > 0.0f ? a : 0.2f * a;  // leaky relu
    aw[(size_t)e * NHH + hh] = a;
    atomicMaxF(&m[d * NHH + hh], a);
  }
}

__global__ void k_gedge2(const int* __restrict__ dst, float* __restrict__ aw,
                         const float* __restrict__ m, float* __restrict__ den) {
  int e = blockIdx.x * blockDim.x + threadIdx.x;
  if (e >= EE + NN) return;
  int d = (e < EE) ? dst[e] : e - EE;
#pragma unroll
  for (int hh = 0; hh < NHH; ++hh) {
    float ex = expf(aw[(size_t)e * NHH + hh] - m[d * NHH + hh]);
    aw[(size_t)e * NHH + hh] = ex;
    atomicAdd(&den[d * NHH + hh], ex);
  }
}

__global__ void k_gedge3(const int* __restrict__ src, const int* __restrict__ dst,
                         const float* __restrict__ aw, const float* __restrict__ den,
                         const float* __restrict__ h, float* __restrict__ outg, int C) {
  int q = (NHH * C) >> 2;
  long i = (long)blockIdx.x * blockDim.x + threadIdx.x;
  if (i >= (long)(EE + NN) * q) return;
  int e = (int)(i / q);
  int off = ((int)(i - (long)e * q)) << 2;
  int s, d;
  if (e < EE) { s = src[e]; d = dst[e]; } else { s = d = e - EE; }
  int hh = (off >= C) ? 1 : 0;
  float alpha = aw[(size_t)e * NHH + hh] / den[d * NHH + hh];
  const float4 hv = *reinterpret_cast<const float4*>(h + (size_t)s * NHH * C + off);
  float* op = outg + (size_t)d * NHH * C + off;
  atomicAdd(op + 0, hv.x * alpha);
  atomicAdd(op + 1, hv.y * alpha);
  atomicAdd(op + 2, hv.z * alpha);
  atomicAdd(op + 3, hv.w * alpha);
}

__global__ void k_gfin(const float* __restrict__ outg, const float* __restrict__ gb,
                       float* __restrict__ out, int C) {
  long i = (long)blockIdx.x * blockDim.x + threadIdx.x;
  if (i >= (long)NN * C) return;
  int o = (int)(i % C);
  long n = i / C;
  out[i] = 0.5f * (outg[n * NHH * C + o] + outg[n * NHH * C + C + o]) + gb[o];
}

// ---------------- final combine ----------------
__global__ void k_final(const float* __restrict__ x1f, const float* __restrict__ x2f,
                        float* __restrict__ out) {
  int n = blockIdx.x * blockDim.x + threadIdx.x;
  if (n >= NN) return;
  float mn = 0.0f;
#pragma unroll
  for (int c = 0; c < 16; ++c) mn += x2f[(size_t)n * 16 + c];
  mn *= (1.0f / 16.0f);
#pragma unroll
  for (int c = 0; c < 16; ++c)
    out[(size_t)n * 16 + c] = tanhf(x1f[(size_t)n * 16 + c] + mn);
}

// ---------------- host orchestration ----------------
static void run_rgcn(hipStream_t st, const h16* xin, const int* src, const int* dst,
                     const int* et, const float* rb, const float* rc, const float* rr,
                     const float* rbias, int dout, h16* Wt, float* H, float* cnt, float* out) {
  int hc = (RRk + 1) * dout;
  k_wrgcn<<<gdiv((long)hc * 64, 256), 256, 0, st>>>(rb, rc, rr, Wt, dout);
  k_gemm<<<dim3(hc / 16, NN / 16), 32, 0, st>>>(xin, Wt, H, hc);
  k_fill<<<gdiv((long)NN * RRk, 256), 256, 0, st>>>(cnt, 0.0f, (long)NN * RRk);
  k_rcount<<<gdiv(EE, 256), 256, 0, st>>>(dst, et, cnt);
  k_rinit<<<gdiv((long)NN * dout, 256), 256, 0, st>>>(H, rbias, out, dout);
  k_rscatter<<<gdiv((long)EE * (dout / 4), 256), 256, 0, st>>>(H, src, dst, et, cnt, out, dout);
}

static void run_gat(hipStream_t st, const h16* xin, const int* src, const int* dst,
                    const float* gw, const float* gas, const float* gad, const float* gb,
                    int C, h16* Wt, float* scratch, float* as_, float* ad_, float* m_,
                    float* den_, float* out) {
  int nhc = NHH * C;
  float* hg   = scratch;                       // [NN, nhc]
  float* outg = scratch + (size_t)NN * 128;    // [NN, nhc]
  float* aw   = scratch + (size_t)NN * 256;    // [(EE+NN), NHH]
  k_wgat<<<gdiv((long)nhc * 64, 256), 256, 0, st>>>(gw, Wt, nhc);
  k_gemm<<<dim3(nhc / 16, NN / 16), 32, 0, st>>>(xin, Wt, hg, nhc);
  k_glogit<<<gdiv((long)NN * NHH, 256), 256, 0, st>>>(hg, gas, gad, as_, ad_, C);
  k_fill<<<gdiv((long)NN * NHH, 256), 256, 0, st>>>(m_, -1e30f, (long)NN * NHH);
  k_fill<<<gdiv((long)NN * NHH, 256), 256, 0, st>>>(den_, 0.0f, (long)NN * NHH);
  k_fill<<<gdiv((long)NN * nhc, 256), 256, 0, st>>>(outg, 0.0f, (long)NN * nhc);
  k_gedge1<<<gdiv(EE + NN, 256), 256, 0, st>>>(src, dst, as_, ad_, aw, m_);
  k_gedge2<<<gdiv(EE + NN, 256), 256, 0, st>>>(dst, aw, m_, den_);
  k_gedge3<<<gdiv((long)(EE + NN) * (nhc / 4), 256), 256, 0, st>>>(src, dst, aw, den_, hg, outg, C);
  k_gfin<<<gdiv((long)NN * C, 256), 256, 0, st>>>(outg, gb, out, C);
}

extern "C" void kernel_launch(void* const* d_in, const int* in_sizes, int n_in,
                              void* d_out, int out_size, void* d_ws, size_t ws_size,
                              hipStream_t stream) {
  (void)in_sizes; (void)n_in; (void)out_size; (void)ws_size;
  const float* x  = (const float*)d_in[0];
  const int* ei   = (const int*)d_in[1];
  const int* et   = (const int*)d_in[2];
  const int* src  = ei;
  const int* dst  = ei + EE;
  const float* P[3][8];
  for (int k = 0; k < 3; ++k)
    for (int j = 0; j < 8; ++j)
      P[k][j] = (const float*)d_in[3 + 8 * k + j];
  // j: 0=rb 1=rc 2=rr 3=rbias 4=gw 5=gas 6=gad 7=gb

  char* w = (char*)d_ws;
  auto take = [&](size_t bytes) -> void* {
    void* p = (void*)w;
    w += (bytes + 255) & ~(size_t)255;
    return p;
  };
  h16* hx   = (h16*)take((size_t)NN * 64 * sizeof(h16));
  h16* hA   = (h16*)take((size_t)NN * 64 * sizeof(h16));
  h16* hB   = (h16*)take((size_t)NN * 64 * sizeof(h16));
  h16* WtR  = (h16*)take((size_t)576 * 64 * sizeof(h16));
  h16* WtG  = (h16*)take((size_t)128 * 64 * sizeof(h16));
  float* BIG = (float*)take((size_t)NN * 576 * sizeof(float));  // H / GAT scratch
  float* cnt = (float*)take((size_t)NN * RRk * sizeof(float));
  float* x1a = (float*)take((size_t)NN * 64 * sizeof(float));
  float* x2a = (float*)take((size_t)NN * 64 * sizeof(float));
  float* x1b = (float*)take((size_t)NN * 64 * sizeof(float));
  float* x2b = (float*)take((size_t)NN * 64 * sizeof(float));
  float* f1  = (float*)take((size_t)NN * 16 * sizeof(float));
  float* f2  = (float*)take((size_t)NN * 16 * sizeof(float));
  float* as_ = (float*)take((size_t)NN * NHH * sizeof(float));
  float* ad_ = (float*)take((size_t)NN * NHH * sizeof(float));
  float* m_  = (float*)take((size_t)NN * NHH * sizeof(float));
  float* dn_ = (float*)take((size_t)NN * NHH * sizeof(float));

  // ---- stage 0: both branches read x ----
  k_cvt<<<gdiv((long)NN * 64, 256), 256, 0, stream>>>(x, hx, (long)NN * 64, 0);
  run_rgcn(stream, hx, src, dst, et, P[0][0], P[0][1], P[0][2], P[0][3], 64, WtR, BIG, cnt, x1a);
  run_gat(stream, hx, src, dst, P[0][4], P[0][5], P[0][6], P[0][7], 64, WtG, BIG, as_, ad_, m_, dn_, x2a);

  // ---- stage 1: relu(x1a) -> rgcn1, relu(x2a) -> gat1 ----
  k_cvt<<<gdiv((long)NN * 64, 256), 256, 0, stream>>>(x1a, hA, (long)NN * 64, 1);
  k_cvt<<<gdiv((long)NN * 64, 256), 256, 0, stream>>>(x2a, hB, (long)NN * 64, 1);
  run_rgcn(stream, hA, src, dst, et, P[1][0], P[1][1], P[1][2], P[1][3], 64, WtR, BIG, cnt, x1b);
  run_gat(stream, hB, src, dst, P[1][4], P[1][5], P[1][6], P[1][7], 64, WtG, BIG, as_, ad_, m_, dn_, x2b);

  // ---- stage 2 (branch swap): rgcn2(relu(x2b)), gat2(relu(x1b)) ----
  k_cvt<<<gdiv((long)NN * 64, 256), 256, 0, stream>>>(x2b, hA, (long)NN * 64, 1);
  k_cvt<<<gdiv((long)NN * 64, 256), 256, 0, stream>>>(x1b, hB, (long)NN * 64, 1);
  run_rgcn(stream, hA, src, dst, et, P[2][0], P[2][1], P[2][2], P[2][3], 16, WtR, BIG, cnt, f1);
  run_gat(stream, hB, src, dst, P[2][4], P[2][5], P[2][6], P[2][7], 16, WtG, BIG, as_, ad_, m_, dn_, f2);

  // ---- final: tanh(x1 + mean(x2)) ----
  k_final<<<gdiv(NN, 256), 256, 0, stream>>>(f1, f2, (float*)d_out);
}
